// RegionProposalNetwork_49177375539820
// MI455X (gfx1250) — matrix operand order repaired and checked by reading, not compiled, not run on hardware
//
#include <hip/hip_runtime.h>
#include <math.h>

// ---------------------------------------------------------------------------
// Types for CDNA5 WMMA (wave32)
// ---------------------------------------------------------------------------
typedef __bf16  bf16;
typedef bf16    v16bf __attribute__((ext_vector_type(16)));
typedef float   v8f   __attribute__((ext_vector_type(8)));
typedef unsigned short ushort_t;

union FragAB {
    v16bf    bf;
    ushort_t u[16];
    uint4    q[2];
};

// ---------------------------------------------------------------------------
// Problem constants
// ---------------------------------------------------------------------------
#define CIN   512
#define COUT  512
#define FH    128
#define FW    128
#define NPIX  (FH * FW)          // 16384
#define PADH  130
#define PADW  132                // padded row stride
#define NANCH (NPIX * 9)         // 147456
#define HM    48                 // head rows padded 45 -> 48
#define PRE_NMS  1000
#define POST_NMS 300
#define CANDMAX  4096
#define NBLK_A   ((NANCH + 255) / 256)   // 576
#define KCH   256                // conv K chunk staged in LDS (32 KB panel)
#define BBOX_CLAMP 4.135166556742356f    // log(1000/16)

__device__ __forceinline__ ushort_t f2bf(float f) {
    unsigned u = __float_as_uint(f);
    u += 0x7FFFu + ((u >> 16) & 1u);     // RNE
    return (ushort_t)(u >> 16);
}
__device__ __forceinline__ float bf2f(ushort_t h) {
    return __uint_as_float(((unsigned)h) << 16);
}
__device__ __forceinline__ unsigned fkey(float f) {
    unsigned u = __float_as_uint(f);
    return (u & 0x80000000u) ? ~u : (u | 0x80000000u);
}
__device__ __forceinline__ bool gt_pair(float sa, int ia, float sb, int ib) {
    return (sa > sb) || (sa == sb && ia < ib);
}

// ---------------------------------------------------------------------------
// Prep kernels
// ---------------------------------------------------------------------------
__global__ void zero_kernel(unsigned* hist, unsigned* sel) {
    int i = threadIdx.x;
    if (i < 256) hist[i] = 0u;
    if (i < 4)   sel[i]  = 0u;
}

// rpn_w (co,ci,3,3) f32 -> Wb[tap][co][ci] bf16
__global__ void prep_convw_kernel(const float* __restrict__ w, ushort_t* __restrict__ Wb) {
    int i = blockIdx.x * 256 + threadIdx.x;
    if (i >= 9 * COUT * CIN) return;
    int t  = i / (COUT * CIN);
    int r  = i % (COUT * CIN);
    int co = r / CIN, ci = r % CIN;
    Wb[i] = f2bf(w[((size_t)co * CIN + ci) * 9 + t]);
}

// cls_w (9,512) + bbox_w (36,512) f32 -> HWb[48][512] bf16 (rows 45..47 zero)
__global__ void prep_headw_kernel(const float* __restrict__ cw, const float* __restrict__ bw,
                                  ushort_t* __restrict__ HWb) {
    int i = blockIdx.x * 256 + threadIdx.x;
    if (i >= HM * CIN) return;
    int m = i / CIN, k = i % CIN;
    float v = 0.0f;
    if (m < 9)       v = cw[m * CIN + k];
    else if (m < 45) v = bw[(m - 9) * CIN + k];
    HWb[i] = f2bf(v);
}

// feat (512,128,128) f32 -> Xpad[512][130][132] bf16, zero halo
__global__ void prep_xpad_kernel(const float* __restrict__ feat, ushort_t* __restrict__ Xpad) {
    int i = blockIdx.x * 256 + threadIdx.x;
    if (i >= CIN * PADH * PADW) return;
    int c = i / (PADH * PADW);
    int r = (i / PADW) % PADH;
    int col = i % PADW;
    float v = 0.0f;
    if (r >= 1 && r <= FH && col >= 1 && col <= FW)
        v = feat[(size_t)c * NPIX + (r - 1) * FW + (col - 1)];
    Xpad[i] = f2bf(v);
}

// ---------------------------------------------------------------------------
// 3x3 conv + ReLU via v_wmma_f32_16x16x32_bf16 (implicit GEMM, 9 taps).
// Block: 256 thr (8 waves). Block tile: M=64 cout x N=256 px (2 rows).
// Wave tile: M=32 x N=64 -> 8 wmma per K-step, 1 wmma per global b128.
// A panel (64 x 256 K) staged in 32 KB LDS; inner 8 K-steps barrier-free.
// ---------------------------------------------------------------------------
__global__ __launch_bounds__(256)
void conv3x3_wmma_kernel(const ushort_t* __restrict__ Wb,
                         const ushort_t* __restrict__ Xpad,
                         const float*    __restrict__ rpn_b,
                         ushort_t*       __restrict__ Xh) {
    __shared__ __align__(16) ushort_t Apanel[64 * KCH];   // 32 KB

    const int tid   = threadIdx.x;
    const int lane  = tid & 31;
    const int wv    = tid >> 5;
    const int l     = lane & 15;
    const int hi    = lane >> 4;
    const int mwave = wv & 1;          // 2 x M32
    const int nwave = wv >> 1;         // 4 x N64
    const int co0   = blockIdx.x * 64;
    const int y0    = blockIdx.y * 2;
    const int yrow  = y0 + (nwave >> 1);
    const int xbase = (nwave & 1) * 64;

    v8f acc[2][4] = {};

    for (int tap = 0; tap < 9; ++tap) {
        const int dy = tap / 3, dx = tap % 3;
        const ushort_t* wbase = Wb + (size_t)tap * COUT * CIN;
        for (int kc = 0; kc < CIN; kc += KCH) {
            __syncthreads();
            // stage A panel: 64 rows x KCH ci = 2048 uint4, 8 per thread
            for (int c = tid; c < 64 * (KCH / 8); c += 256) {
                int row = c >> 5;              // KCH/8 = 32 uint4 per row
                int seg = c & 31;
                *((uint4*)(Apanel + row * KCH + seg * 8)) =
                    *((const uint4*)(wbase + (size_t)(co0 + row) * CIN + kc + seg * 8));
            }
            __syncthreads();

            for (int kb = 0; kb < KCH; kb += 32) {
                FragAB fa[2];   // A 16x32 per ISA layout: two aligned ds b128
                #pragma unroll
                for (int mt = 0; mt < 2; ++mt) {
                    const uint4* ap = (const uint4*)(Apanel +
                        (mwave * 32 + mt * 16 + l) * KCH + kb + hi * 8);
                    fa[mt].q[0] = ap[0];   // K = hi*8 + 0..7
                    fa[mt].q[1] = ap[2];   // K = 16 + hi*8 + 0..7
                }
                // B 32x16: lane holds column N = l, K in ISA table order
                const ushort_t* brow = Xpad + (size_t)(kc + kb + hi * 8) * (PADH * PADW)
                                     + (yrow + dy) * PADW + xbase + dx + l;
                #pragma unroll
                for (int nt = 0; nt < 4; ++nt) {
                    FragAB fb;
                    #pragma unroll
                    for (int s = 0; s < 16; ++s)
                        fb.u[s] = brow[(size_t)((s & 7) + (s >> 3) * 16) * (PADH * PADW) + nt * 16];
                    #pragma unroll
                    for (int mt = 0; mt < 2; ++mt)
                        acc[mt][nt] = __builtin_amdgcn_wmma_f32_16x16x32_bf16(
                            false, fa[mt].bf, false, fb.bf, (short)0,
                            acc[mt][nt], false, false);
                }
            }
        }
    }
    // Epilogue: bias + ReLU -> bf16 hidden activation [co][pixel]
    #pragma unroll
    for (int mt = 0; mt < 2; ++mt)
        #pragma unroll
        for (int nt = 0; nt < 4; ++nt)
            #pragma unroll
            for (int r = 0; r < 8; ++r) {
                int co = co0 + mwave * 32 + mt * 16 + hi * 8 + r;
                int x  = xbase + nt * 16 + l;
                float v = acc[mt][nt][r] + rpn_b[co];
                v = fmaxf(v, 0.0f);
                Xh[(size_t)co * NPIX + yrow * FW + x] = f2bf(v);
            }
}

// ---------------------------------------------------------------------------
// Heads as WMMA GEMM: M=48 (cls 9 + bbox 36 + pad), K=512, N=16384.
// Block: 256 thr (8 waves), each wave one 16-px column tile, 3 M-tiles.
// Weights staged once in 48 KB LDS; B rows of Xh are 32B-aligned.
// ---------------------------------------------------------------------------
__global__ __launch_bounds__(256)
void heads_wmma_kernel(const ushort_t* __restrict__ Xh,
                       const ushort_t* __restrict__ HWb,
                       float* __restrict__ Hout) {
    __shared__ __align__(16) ushort_t Wpanel[HM * CIN];   // 48 KB

    const int tid  = threadIdx.x;
    const int lane = tid & 31;
    const int wv   = tid >> 5;
    const int l    = lane & 15;
    const int hi   = lane >> 4;
    const int pbase = blockIdx.x * 128 + wv * 16;

    for (int c = tid; c < HM * (CIN / 8); c += 256)
        *((uint4*)(Wpanel + c * 8)) = *((const uint4*)(HWb + c * 8));
    __syncthreads();

    v8f acc[3] = {};
    for (int kb = 0; kb < CIN; kb += 32) {
        FragAB fb;   // B: column N = pbase+l, K in ISA table order
        {
            const ushort_t* bp = Xh + (size_t)(kb + hi * 8) * NPIX + pbase + l;
            #pragma unroll
            for (int s = 0; s < 16; ++s)
                fb.u[s] = bp[(size_t)((s & 7) + (s >> 3) * 16) * NPIX];
        }
        #pragma unroll
        for (int mt = 0; mt < 3; ++mt) {
            FragAB fa;
            const uint4* ap = (const uint4*)(Wpanel + (mt * 16 + l) * CIN + kb + hi * 8);
            fa.q[0] = ap[0];
            fa.q[1] = ap[2];
            acc[mt] = __builtin_amdgcn_wmma_f32_16x16x32_bf16(
                false, fa.bf, false, fb.bf, (short)0, acc[mt], false, false);
        }
    }
    #pragma unroll
    for (int mt = 0; mt < 3; ++mt)
        #pragma unroll
        for (int r = 0; r < 8; ++r) {
            int m = mt * 16 + hi * 8 + r;
            Hout[(size_t)m * NPIX + pbase + l] = acc[mt][r];
        }
}

// ---------------------------------------------------------------------------
// Anchors + apply_deltas per pixel (reads 45 coalesced rows of Hout)
// ---------------------------------------------------------------------------
__global__ __launch_bounds__(256)
void anchor_kernel(const float* __restrict__ Hout,
                   const float* __restrict__ cls_b, const float* __restrict__ bbox_b,
                   const int* __restrict__ ihp, const int* __restrict__ iwp,
                   float* __restrict__ scores_all, float4* __restrict__ boxes_all) {
    const int p = blockIdx.x * 256 + threadIdx.x;
    if (p >= NPIX) return;
    float acc[45];
    #pragma unroll
    for (int m = 0; m < 45; ++m) acc[m] = Hout[(size_t)m * NPIX + p];

    const float sx = (float)iwp[0] / (float)FW;   // 16
    const float sy = (float)ihp[0] / (float)FH;   // 16
    const int px = p % FW, py = p / FW;
    const float xx = px * sx, yy = py * sy;
    const float scales[3] = {128.0f, 256.0f, 512.0f};
    const float ratios[3] = {0.5f, 1.0f, 2.0f};

    #pragma unroll
    for (int a = 0; a < 9; ++a) {
        scores_all[(size_t)p * 9 + a] = acc[a] + cls_b[a];
        int ar = a / 3, si = a % 3;
        float hr = sqrtf(ratios[ar]);
        float wa = (1.0f / hr) * scales[si];
        float ha = hr * scales[si];
        float dxv = acc[9 + a * 4 + 0] + bbox_b[a * 4 + 0];
        float dyv = acc[9 + a * 4 + 1] + bbox_b[a * 4 + 1];
        float dwv = fminf(acc[9 + a * 4 + 2] + bbox_b[a * 4 + 2], BBOX_CLAMP);
        float dhv = fminf(acc[9 + a * 4 + 3] + bbox_b[a * 4 + 3], BBOX_CLAMP);
        float pcx = dxv * wa + xx, pcy = dyv * ha + yy;
        float pw = __expf(dwv) * wa, ph = __expf(dhv) * ha;
        boxes_all[(size_t)p * 9 + a] =
            make_float4(pcx - 0.5f * pw, pcy - 0.5f * ph, pcx + 0.5f * pw, pcy + 0.5f * ph);
    }
}

// ---------------------------------------------------------------------------
// Top-k candidate selection: radix-bin threshold + ordered compaction
// ---------------------------------------------------------------------------
__global__ void hist_kernel(const float* __restrict__ s, unsigned* __restrict__ hist) {
    int i = blockIdx.x * 256 + threadIdx.x;
    if (i < NANCH) atomicAdd(&hist[fkey(s[i]) >> 24], 1u);
}

__global__ void select_bin_kernel(const unsigned* __restrict__ hist, unsigned* __restrict__ sel) {
    if (threadIdx.x == 0) {
        unsigned cum = 0; int b = 255;
        for (; b > 0; --b) { cum += hist[b]; if (cum >= (unsigned)PRE_NMS) break; }
        sel[0] = (unsigned)b;
    }
}

__global__ void count_kernel(const float* __restrict__ s, const unsigned* __restrict__ sel,
                             unsigned* __restrict__ blkcnt) {
    __shared__ unsigned cnt;
    if (threadIdx.x == 0) cnt = 0u;
    __syncthreads();
    int i = blockIdx.x * 256 + threadIdx.x;
    if (i < NANCH && (fkey(s[i]) >> 24) >= sel[0]) atomicAdd(&cnt, 1u);
    __syncthreads();
    if (threadIdx.x == 0) blkcnt[blockIdx.x] = cnt;
}

__global__ void scan_kernel(const unsigned* __restrict__ blkcnt, unsigned* __restrict__ blkoff,
                            unsigned* __restrict__ sel) {
    if (threadIdx.x == 0) {
        unsigned run = 0;
        for (int i = 0; i < NBLK_A; ++i) { blkoff[i] = run; run += blkcnt[i]; }
        sel[1] = run;
    }
}

__global__ void compact_kernel(const float* __restrict__ s, const unsigned* __restrict__ sel,
                               const unsigned* __restrict__ blkoff,
                               float* __restrict__ cand_s, int* __restrict__ cand_i) {
    __shared__ unsigned scn[256];
    int tid = threadIdx.x;
    int i = blockIdx.x * 256 + tid;
    unsigned pred = (i < NANCH && (fkey(s[i]) >> 24) >= sel[0]) ? 1u : 0u;
    scn[tid] = pred;
    __syncthreads();
    for (int off = 1; off < 256; off <<= 1) {
        unsigned add = (tid >= off) ? scn[tid - off] : 0u;
        __syncthreads();
        scn[tid] += add;
        __syncthreads();
    }
    if (pred) {
        unsigned pos = blkoff[blockIdx.x] + scn[tid] - 1u;  // ordered, deterministic
        if (pos < (unsigned)CANDMAX) { cand_s[pos] = s[i]; cand_i[pos] = i; }
    }
}

// ---------------------------------------------------------------------------
// Finalize: sort candidates, clip, min-size, NMS, top-300, sigmoid
// ---------------------------------------------------------------------------
__device__ void sort1024(float* ss, int* si, int tid) {
    for (int k = 2; k <= 1024; k <<= 1)
        for (int j = k >> 1; j > 0; j >>= 1) {
            __syncthreads();
            int ixj = tid ^ j;
            if (ixj > tid) {
                float sa = ss[tid], sb = ss[ixj];
                int ia = si[tid], ib = si[ixj];
                bool up = ((tid & k) == 0);
                bool sw = up ? gt_pair(sb, ib, sa, ia) : gt_pair(sa, ia, sb, ib);
                if (sw) { ss[tid] = sb; si[tid] = ib; ss[ixj] = sa; si[ixj] = ia; }
            }
        }
    __syncthreads();
}

__global__ __launch_bounds__(1024)
void finalize_kernel(const float* __restrict__ cand_s, const int* __restrict__ cand_i,
                     const unsigned* __restrict__ sel, const float4* __restrict__ boxes_all,
                     const int* __restrict__ ihp, const int* __restrict__ iwp,
                     float* __restrict__ out) {
    __shared__ float ss[CANDMAX];
    __shared__ int   si[CANDMAX];
    __shared__ float bx1[1024], by1[1024], bx2[1024], by2[1024];
    __shared__ unsigned keep[1024];
    const int tid = threadIdx.x;
    const float ih = (float)ihp[0], iw = (float)iwp[0];

    int M = (int)sel[1]; if (M > CANDMAX) M = CANDMAX;
    for (int i = tid; i < CANDMAX; i += 1024) {
        if (i < M) { ss[i] = cand_s[i]; si[i] = cand_i[i]; }
        else       { ss[i] = -3.0e38f;  si[i] = 0x7fffffff; }
    }
    __syncthreads();

    // bitonic sort 4096 (desc)
    for (int k = 2; k <= CANDMAX; k <<= 1)
        for (int j = k >> 1; j > 0; j >>= 1) {
            __syncthreads();
            for (int v = tid; v < CANDMAX; v += 1024) {
                int ixj = v ^ j;
                if (ixj > v) {
                    float sa = ss[v], sb = ss[ixj];
                    int ia = si[v], ib = si[ixj];
                    bool up = ((v & k) == 0);
                    bool sw = up ? gt_pair(sb, ib, sa, ia) : gt_pair(sa, ia, sb, ib);
                    if (sw) { ss[v] = sb; si[v] = ib; ss[ixj] = sa; si[ixj] = ia; }
                }
            }
        }
    __syncthreads();

    // top-1000: clip to image, min-size filter
    float x1 = 0, y1 = 0, x2 = 0, y2 = 0, sc = -1.0e9f;
    if (tid < PRE_NMS && tid < M && ss[tid] > -1.0e37f) {
        float4 b = boxes_all[si[tid]];
        x1 = fminf(fmaxf(b.x, 0.0f), iw);
        y1 = fminf(fmaxf(b.y, 0.0f), ih);
        x2 = fminf(fmaxf(b.z, 0.0f), iw);
        y2 = fminf(fmaxf(b.w, 0.0f), ih);
        sc = ss[tid];
        if ((x2 - x1 < 16.0f) || (y2 - y1 < 16.0f)) sc = -1.0e9f;
    }
    __syncthreads();
    bx1[tid] = x1; by1[tid] = y1; bx2[tid] = x2; by2[tid] = y2;
    ss[tid] = sc; si[tid] = tid;
    __syncthreads();

    sort1024(ss, si, tid);           // second top_k
    {   // permute boxes to new order
        int j = si[tid];
        float tx1 = bx1[j], ty1 = by1[j], tx2 = bx2[j], ty2 = by2[j];
        __syncthreads();
        bx1[tid] = tx1; by1[tid] = ty1; bx2[tid] = tx2; by2[tid] = ty2;
        __syncthreads();
    }

    keep[tid] = (ss[tid] > -1.0e8f) ? 1u : 0u;
    float myArea = (bx2[tid] - bx1[tid]) * (by2[tid] - by1[tid]);
    __syncthreads();

    for (int i = 0; i < PRE_NMS; ++i) {       // greedy NMS, IoU > 0.7
        float ix1 = bx1[i], iy1 = by1[i], ix2 = bx2[i], iy2 = by2[i];
        float ia = (ix2 - ix1) * (iy2 - iy1);
        bool ki = (keep[i] != 0u);
        __syncthreads();
        if (ki && tid > i && keep[tid]) {
            float xx1 = fmaxf(ix1, bx1[tid]), yy1 = fmaxf(iy1, by1[tid]);
            float xx2 = fminf(ix2, bx2[tid]), yy2 = fminf(iy2, by2[tid]);
            float inter = fmaxf(xx2 - xx1, 0.0f) * fmaxf(yy2 - yy1, 0.0f);
            float iou = inter / (ia + myArea - inter + 1e-9f);
            if (iou > 0.7f) keep[tid] = 0u;
        }
        __syncthreads();
    }

    float s3 = keep[tid] ? ss[tid] : -1.0e9f;
    __syncthreads();
    ss[tid] = s3; si[tid] = tid;
    __syncthreads();
    sort1024(ss, si, tid);            // final top-300

    if (tid < POST_NMS) {
        int j = si[tid];
        out[tid * 4 + 0] = bx1[j];
        out[tid * 4 + 1] = by1[j];
        out[tid * 4 + 2] = bx2[j];
        out[tid * 4 + 3] = by2[j];
        out[4 * POST_NMS + tid] = 1.0f / (1.0f + __expf(-ss[tid]));
    }
}

// ---------------------------------------------------------------------------
// Host launcher
// ---------------------------------------------------------------------------
extern "C" void kernel_launch(void* const* d_in, const int* in_sizes, int n_in,
                              void* d_out, int out_size, void* d_ws, size_t ws_size,
                              hipStream_t stream) {
    const float* feat   = (const float*)d_in[0];
    const float* rpn_w  = (const float*)d_in[1];
    const float* rpn_b  = (const float*)d_in[2];
    const float* cls_w  = (const float*)d_in[3];
    const float* cls_b  = (const float*)d_in[4];
    const float* bbox_w = (const float*)d_in[5];
    const float* bbox_b = (const float*)d_in[6];
    const int*   ih     = (const int*)d_in[7];
    const int*   iw     = (const int*)d_in[8];
    float* out = (float*)d_out;

    char* ws = (char*)d_ws;
    size_t off = 0;
    auto take = [&](size_t bytes) { char* p = ws + off; off += (bytes + 255) & ~(size_t)255; return p; };
    ushort_t* Wb       = (ushort_t*)take((size_t)9 * COUT * CIN * 2);
    ushort_t* Xpad     = (ushort_t*)take((size_t)CIN * PADH * PADW * 2);
    ushort_t* Xh       = (ushort_t*)take((size_t)COUT * NPIX * 2);
    ushort_t* HWb      = (ushort_t*)take((size_t)HM * CIN * 2);
    float*    Hout     = (float*)   take((size_t)HM * NPIX * 4);
    float*    scoresA  = (float*)   take((size_t)NANCH * 4);
    float4*   boxesA   = (float4*)  take((size_t)NANCH * 16);
    unsigned* hist     = (unsigned*)take(256 * 4);
    unsigned* sel      = (unsigned*)take(4 * 4);
    unsigned* blkcnt   = (unsigned*)take((size_t)NBLK_A * 4);
    unsigned* blkoff   = (unsigned*)take((size_t)NBLK_A * 4);
    float*    cand_s   = (float*)   take((size_t)CANDMAX * 4);
    int*      cand_i   = (int*)     take((size_t)CANDMAX * 4);

    zero_kernel<<<1, 256, 0, stream>>>(hist, sel);
    prep_convw_kernel<<<(9 * COUT * CIN + 255) / 256, 256, 0, stream>>>(rpn_w, Wb);
    prep_headw_kernel<<<(HM * CIN + 255) / 256, 256, 0, stream>>>(cls_w, bbox_w, HWb);
    prep_xpad_kernel<<<(CIN * PADH * PADW + 255) / 256, 256, 0, stream>>>(feat, Xpad);

    conv3x3_wmma_kernel<<<dim3(COUT / 64, FH / 2), 256, 0, stream>>>(Wb, Xpad, rpn_b, Xh);

    heads_wmma_kernel<<<NPIX / 128, 256, 0, stream>>>(Xh, HWb, Hout);
    anchor_kernel<<<NPIX / 256, 256, 0, stream>>>(Hout, cls_b, bbox_b, ih, iw, scoresA, boxesA);

    hist_kernel<<<NBLK_A, 256, 0, stream>>>(scoresA, hist);
    select_bin_kernel<<<1, 32, 0, stream>>>(hist, sel);
    count_kernel<<<NBLK_A, 256, 0, stream>>>(scoresA, sel, blkcnt);
    scan_kernel<<<1, 32, 0, stream>>>(blkcnt, blkoff, sel);
    compact_kernel<<<NBLK_A, 256, 0, stream>>>(scoresA, sel, blkoff, cand_s, cand_i);
    finalize_kernel<<<1, 1024, 0, stream>>>(cand_s, cand_i, sel, boxesA, ih, iw, out);
}
